// self_attention_15169824490061
// MI455X (gfx1250) — compile-verified, matrix operand order checked by reading, not run on hardware
//
#include <hip/hip_runtime.h>
#include <hip/hip_bf16.h>

typedef __attribute__((ext_vector_type(16))) _Float16 v16h;
typedef __attribute__((ext_vector_type(8)))  _Float16 v8h;
typedef __attribute__((ext_vector_type(8)))  float    v8f;
typedef int v4i __attribute__((vector_size(16)));

#define BQ 2
#define SQ 2048
#define DQ 512
#define HQ 8
#define DH 64
#define ROWS (BQ * SQ)          // 4096
#define D3  (3 * DQ)            // 1536

#define QW    2                 // q-tiles per wave in attention
#define WAVES 4                 // waves per attention block
#define QBLK  (16 * QW * WAVES) // 128 q rows per block

__device__ __forceinline__ v8f wmma_f16(v16h a, v16h b, v8f c) {
  return __builtin_amdgcn_wmma_f32_16x16x32_f16(
      /*neg_a=*/false, a, /*neg_b=*/false, b,
      /*c_mod=*/(short)0, c, /*reuse_a=*/false, /*reuse_b=*/false);
}

// Load one 16x32 f16 WMMA fragment (A-layout; also B-layout with lane<->N)
// from a row-major matrix `base` with leading dimension `ld` (halves),
// starting at K-column `kb`.  ISA 7.12.2 (16-bit A 16x32):
// lanes 0-15 : M=lane,    elems 0..7 = K kb+0..7,  elems 8..15 = K kb+16..23
// lanes 16-31: M=lane-16, elems 0..7 = K kb+8..15, elems 8..15 = K kb+24..31
__device__ __forceinline__ v16h load_frag16(const _Float16* __restrict__ base,
                                            int ld, int kb) {
  const int lane = threadIdx.x & 31;
  const int row  = lane & 15;
  const int hi   = lane >> 4;
  const _Float16* p = base + row * ld + kb + hi * 8;
  v8h lo = *(const v8h*)(p);
  v8h up = *(const v8h*)(p + 16);
  v16h r;
#pragma unroll
  for (int i = 0; i < 8; ++i) { r[i] = lo[i]; r[i + 8] = up[i]; }
  return r;
}

// 16-byte global -> LDS copy; async (ASYNCcnt / bypasses VGPRs) when the
// gfx1250 builtin is available, register bounce otherwise.
__device__ __forceinline__ void copy16_g2l(_Float16* ldst, const _Float16* gsrc) {
#if __has_builtin(__builtin_amdgcn_global_load_async_to_lds_b128)
  __builtin_amdgcn_global_load_async_to_lds_b128(
      (__attribute__((address_space(1))) v4i*)gsrc,
      (__attribute__((address_space(3))) v4i*)ldst,
      /*imm_offset=*/0, /*cpol=*/0);
#else
  *(v8h*)ldst = *(const v8h*)gsrc;
#endif
}

__device__ __forceinline__ void wait_async0() {
#if __has_builtin(__builtin_amdgcn_s_wait_asynccnt)
  __builtin_amdgcn_s_wait_asynccnt(0);
#else
  asm volatile("s_wait_asynccnt 0x0" ::: "memory");
#endif
}

__device__ __forceinline__ void wait_ds0() {
  asm volatile("s_wait_dscnt 0x0" ::: "memory");
}

// ---------------------------------------------------------------- converts
__global__ void cvt_f32_to_f16_kernel(const float* __restrict__ in,
                                      _Float16* __restrict__ out, int n) {
  int i = blockIdx.x * blockDim.x + threadIdx.x;
  if (i < n) out[i] = (_Float16)in[i];
}

// ---------------------------------------------------------------- in_proj
// qvk = x @ w_in^T + b_in ; split (reference quirk) as q, v, k.
// Writes Q/K head-major [bh][s][dh] and V transposed [bh][dh][s], all f16.
__global__ __launch_bounds__(32)
void inproj_kernel(const _Float16* __restrict__ xh,
                   const _Float16* __restrict__ wh,
                   const float* __restrict__ bias,
                   _Float16* __restrict__ Qh,
                   _Float16* __restrict__ Kh,
                   _Float16* __restrict__ Vt) {
  const int rbase = blockIdx.x * 16;   // output row in [0, ROWS)
  const int jbase = blockIdx.y * 64;   // output col in [0, 3D)
  const int lane  = threadIdx.x & 31;
  const int nn = lane & 15, hi = lane >> 4;

  v8f acc[4] = {};
  const _Float16* arow = xh + (size_t)rbase * DQ;
  for (int kb = 0; kb < DQ; kb += 32) {
    v16h a = load_frag16(arow, DQ, kb);
#pragma unroll
    for (int t = 0; t < 4; ++t) {
      v16h bf = load_frag16(wh + (size_t)(jbase + t * 16) * DQ, DQ, kb);
      acc[t] = wmma_f16(a, bf, acc[t]);
    }
  }
#pragma unroll
  for (int t = 0; t < 4; ++t) {
    const int j = jbase + t * 16 + nn;
    const float bj = bias[j];
#pragma unroll
    for (int r = 0; r < 8; ++r) {
      const int row = rbase + r + hi * 8;
      const int b = row >> 11, s = row & (SQ - 1);
      const _Float16 hv = (_Float16)(acc[t][r] + bj);
      if (j < DQ) {                       // Q
        const int h = j >> 6, d = j & 63;
        Qh[(((size_t)(b * HQ + h)) * SQ + s) * DH + d] = hv;
      } else if (j < 2 * DQ) {            // V (second chunk!), transposed
        const int jj = j - DQ, h = jj >> 6, d = jj & 63;
        Vt[(((size_t)(b * HQ + h)) * DH + d) * SQ + s] = hv;
      } else {                            // K
        const int jj = j - 2 * DQ, h = jj >> 6, d = jj & 63;
        Kh[(((size_t)(b * HQ + h)) * SQ + s) * DH + d] = hv;
      }
    }
  }
}

// ---------------------------------------------------------------- attention
// Block = 4 waves, 128 consecutive q rows of one (b,h).  Per 32-key step:
// stage K (32x64) and Vt (64x32) tiles into LDS with async global->LDS
// copies, then each wave reads the shared fragments once and advances its
// two 16-row q-tiles (flash-attention online softmax).
__global__ __launch_bounds__(128)
void attn_kernel(const _Float16* __restrict__ Qh,
                 const _Float16* __restrict__ Kh,
                 const _Float16* __restrict__ Vt,
                 _Float16* __restrict__ ah) {
  __shared__ __align__(16) _Float16 Ktile[32 * DH];        // [key][d]   4 KB
  __shared__ __align__(16) _Float16 Vtile[DH * 32];        // [d][key]   4 KB
  __shared__ __align__(16) _Float16 probs[WAVES][16 * 32]; //            4 KB

  const int tid  = threadIdx.x;
  const int wave = tid >> 5;
  const int lane = tid & 31;
  const int nn = lane & 15, hi = lane >> 4;

  const int h = blockIdx.y, b = blockIdx.z;
  const size_t bh = (size_t)(b * HQ + h);
  const _Float16* Qbase = Qh + bh * SQ * DH;
  const _Float16* Kbase = Kh + bh * SQ * DH;
  const _Float16* Vbase = Vt + bh * DH * SQ;

  const int qg = blockIdx.x * QBLK;        // block's first q row
  const int qw = qg + wave * (16 * QW);    // this wave's first q row

  v16h aq[QW][2];
#pragma unroll
  for (int u = 0; u < QW; ++u) {
    aq[u][0] = load_frag16(Qbase + (size_t)(qw + u * 16) * DH, DH, 0);
    aq[u][1] = load_frag16(Qbase + (size_t)(qw + u * 16) * DH, DH, 32);
  }

  float m[QW][8], l[QW][8];
  v8f o[QW][4] = {};
#pragma unroll
  for (int u = 0; u < QW; ++u)
#pragma unroll
    for (int r = 0; r < 8; ++r) { m[u][r] = -__builtin_inff(); l[u][r] = 0.0f; }

  const float scale = 0.125f;              // 1/sqrt(64)
  const int kend = qg + QBLK;              // last key any wave in block needs

  for (int kb = 0; kb < kend; kb += 32) {
    __syncthreads();                       // previous step's LDS reads done
    // ---- stage K[kb..kb+32) (contiguous 4 KB) and Vt[:, kb..kb+32) ----
#pragma unroll
    for (int it = 0; it < 2; ++it) {
      const int c = tid + it * 128;        // 16-byte chunk id, 0..255
      copy16_g2l(&Ktile[c * 8], Kbase + (size_t)kb * DH + c * 8);
      const int d = c >> 2, of = c & 3;
      copy16_g2l(&Vtile[d * 32 + of * 8],
                 Vbase + (size_t)d * SQ + kb + of * 8);
    }
    wait_async0();
    __syncthreads();

    if (kb < qw + 16 * QW) {               // wave has unmasked work
      // shared fragments: 4 K-frags (2 key sub-tiles x dh halves), 4 V-frags
      v16h bk[2][2], bv[4];
#pragma unroll
      for (int t = 0; t < 2; ++t) {
        bk[t][0] = load_frag16(&Ktile[t * 16 * DH], DH, 0);
        bk[t][1] = load_frag16(&Ktile[t * 16 * DH], DH, 32);
      }
#pragma unroll
      for (int t = 0; t < 4; ++t)
        bv[t] = load_frag16(&Vtile[t * 16 * 32], 32, 0);

#pragma unroll
      for (int u = 0; u < QW; ++u) {
        const int qt = qw + u * 16;
        const int tkend = qt + 16;
        if (kb >= tkend) continue;         // this q-tile fully masked here
        float sc[2][8];
#pragma unroll
        for (int t = 0; t < 2; ++t) {
          const int kt = kb + t * 16;
          if (kt < tkend) {
            v8f s = {};
            s = wmma_f16(aq[u][0], bk[t][0], s);
            s = wmma_f16(aq[u][1], bk[t][1], s);
#pragma unroll
            for (int r = 0; r < 8; ++r) {
              const int row = qt + r + hi * 8;
              const int col = kt + nn;
              sc[t][r] = (col <= row) ? s[r] * scale : -__builtin_inff();
            }
          } else {
#pragma unroll
            for (int r = 0; r < 8; ++r) sc[t][r] = -__builtin_inff();
          }
        }
#pragma unroll
        for (int r = 0; r < 8; ++r) {
          float tm = fmaxf(sc[0][r], sc[1][r]);
          tm = fmaxf(tm, __shfl_xor(tm, 1));
          tm = fmaxf(tm, __shfl_xor(tm, 2));
          tm = fmaxf(tm, __shfl_xor(tm, 4));
          tm = fmaxf(tm, __shfl_xor(tm, 8));
          const float mn = fmaxf(m[u][r], tm);
          const float f  = __expf(m[u][r] - mn);   // 0 on first step
          const float p0 = __expf(sc[0][r] - mn);
          const float p1 = __expf(sc[1][r] - mn);
          float ps = p0 + p1;
          ps += __shfl_xor(ps, 1);
          ps += __shfl_xor(ps, 2);
          ps += __shfl_xor(ps, 4);
          ps += __shfl_xor(ps, 8);
          l[u][r] = l[u][r] * f + ps;
          m[u][r] = mn;
#pragma unroll
          for (int t = 0; t < 4; ++t) o[u][t][r] *= f;
          const int mrow = r + hi * 8;
          probs[wave][mrow * 32 + nn]      = (_Float16)p0;
          probs[wave][mrow * 32 + 16 + nn] = (_Float16)p1;
        }
        wait_ds0();                        // wave-private LDS RAW ordering
        {
          const int koff = hi * 8;
          const v8h plo = *(const v8h*)&probs[wave][nn * 32 + koff];
          const v8h pup = *(const v8h*)&probs[wave][nn * 32 + koff + 16];
          v16h ap;
#pragma unroll
          for (int i = 0; i < 8; ++i) { ap[i] = plo[i]; ap[i + 8] = pup[i]; }
#pragma unroll
          for (int t = 0; t < 4; ++t) o[u][t] = wmma_f16(ap, bv[t], o[u][t]);
        }
      }
    }
  }

  // epilogue: normalize, write back to [B, S, D] (f16 staging for out_proj)
#pragma unroll
  for (int u = 0; u < QW; ++u)
#pragma unroll
    for (int t = 0; t < 4; ++t)
#pragma unroll
      for (int r = 0; r < 8; ++r) {
        const int row = qw + u * 16 + r + hi * 8;
        const int d = t * 16 + nn;
        const float v = o[u][t][r] / l[u][r];
        ah[((size_t)b * SQ + row) * DQ + h * DH + d] = (_Float16)v;
      }
}

// ---------------------------------------------------------------- out_proj
__global__ __launch_bounds__(32)
void outproj_kernel(const _Float16* __restrict__ ah,
                    const _Float16* __restrict__ wh,
                    const float* __restrict__ bias,
                    float* __restrict__ out) {
  const int rbase = blockIdx.x * 16;
  const int jbase = blockIdx.y * 64;
  const int lane  = threadIdx.x & 31;
  const int nn = lane & 15, hi = lane >> 4;

  v8f acc[4] = {};
  const _Float16* arow = ah + (size_t)rbase * DQ;
  for (int kb = 0; kb < DQ; kb += 32) {
    v16h a = load_frag16(arow, DQ, kb);
#pragma unroll
    for (int t = 0; t < 4; ++t) {
      v16h bf = load_frag16(wh + (size_t)(jbase + t * 16) * DQ, DQ, kb);
      acc[t] = wmma_f16(a, bf, acc[t]);
    }
  }
#pragma unroll
  for (int t = 0; t < 4; ++t) {
    const int j = jbase + t * 16 + nn;
    const float bj = bias[j];
#pragma unroll
    for (int r = 0; r < 8; ++r) {
      const int row = rbase + r + hi * 8;
      out[(size_t)row * DQ + j] = acc[t][r] + bj;
    }
  }
}

// ---------------------------------------------------------------- launch
extern "C" void kernel_launch(void* const* d_in, const int* in_sizes, int n_in,
                              void* d_out, int out_size, void* d_ws, size_t ws_size,
                              hipStream_t stream) {
  const float* x     = (const float*)d_in[0];  // [B,S,D]
  const float* w_in  = (const float*)d_in[1];  // [3D,D]
  const float* b_in  = (const float*)d_in[2];  // [3D]
  const float* w_out = (const float*)d_in[3];  // [D,D]
  const float* b_out = (const float*)d_in[4];  // [D]
  (void)in_sizes; (void)n_in; (void)out_size; (void)ws_size;
  float* out = (float*)d_out;

  // workspace carve-up (all f16)
  char* ws = (char*)d_ws;
  _Float16* xh  = (_Float16*)ws;  ws += (size_t)ROWS * DQ * 2;       // 4 MB
  _Float16* wih = (_Float16*)ws;  ws += (size_t)D3 * DQ * 2;         // 1.5 MB
  _Float16* woh = (_Float16*)ws;  ws += (size_t)DQ * DQ * 2;         // 0.5 MB
  _Float16* Qh  = (_Float16*)ws;  ws += (size_t)BQ * HQ * SQ * DH * 2;
  _Float16* Kh  = (_Float16*)ws;  ws += (size_t)BQ * HQ * SQ * DH * 2;
  _Float16* Vt  = (_Float16*)ws;  ws += (size_t)BQ * HQ * SQ * DH * 2;
  _Float16* ah  = (_Float16*)ws;  ws += (size_t)ROWS * DQ * 2;

  const int nx = ROWS * DQ, nwi = D3 * DQ, nwo = DQ * DQ;
  cvt_f32_to_f16_kernel<<<(nx  + 255) / 256, 256, 0, stream>>>(x,     xh,  nx);
  cvt_f32_to_f16_kernel<<<(nwi + 255) / 256, 256, 0, stream>>>(w_in,  wih, nwi);
  cvt_f32_to_f16_kernel<<<(nwo + 255) / 256, 256, 0, stream>>>(w_out, woh, nwo);

  inproj_kernel<<<dim3(ROWS / 16, D3 / 64), 32, 0, stream>>>(
      xh, wih, b_in, Qh, Kh, Vt);

  attn_kernel<<<dim3(SQ * BQ / QBLK / BQ, HQ, BQ), 128, 0, stream>>>(
      Qh, Kh, Vt, ah);

  outproj_kernel<<<dim3(ROWS / 16, DQ / 64), 32, 0, stream>>>(
      ah, woh, b_out, out);
}
